// MAF_27951647162958
// MI455X (gfx1250) — matrix operand order repaired and checked by reading, not compiled, not run on hardware
//
#include <hip/hip_runtime.h>
#include <hip/hip_bf16.h>

// MAF inverse (MADE 64->320->320->128), fully fused per 32-row tile.
// All 64 autoregressive steps run inside one workgroup with masked f16
// weights resident in LDS; matmuls use v_wmma_f32_16x16x32_f16 (wave32).

typedef __attribute__((ext_vector_type(16))) _Float16 v16h;
typedef __attribute__((ext_vector_type(8)))  float    v8f;

#define B_    64
#define T_    1024
#define DIM_  64
#define HID_  320
#define ROWS_ 32      // rows per workgroup
#define NTHR_ 320     // 10 wave32s
#define NWV_  10

union HalfVec { v16h v; uint4 u[2]; _Float16 h[16]; };

// A fragment: 16x32 f16 (MxK), row-major source with leading dim ld (halfs).
// Lane l<16: row M=l, K {0..7, 16..23}; lane>=16: row M=l-16, K {8..15, 24..31}.
__device__ inline v16h load_frag_a(const _Float16* src, int ld, int lane) {
  HalfVec A;
  int m  = lane & 15;
  int kb = (lane >> 4) << 3;            // 0 or 8
  const _Float16* p = src + m * ld + kb;
  A.u[0] = *(const uint4*)(p);          // K = kb .. kb+7
  A.u[1] = *(const uint4*)(p + 16);     // K = kb+16 .. kb+23
  return A.v;
}

// B fragment: 32x16 f16 (KxN). B[k][n] = W[n][k] with W row-major (N x K).
// wsrc points at W[n0][kk]. Lane l<16: col n=l, K {0..7, 16..23}; lane>=16: K {8..15, 24..31}.
__device__ inline v16h load_frag_b(const _Float16* wsrc, int ld, int lane) {
  HalfVec Bv;
  int n  = lane & 15;
  int kb = (lane >> 4) << 3;            // 0 or 8
  const _Float16* p = wsrc + n * ld + kb;
  Bv.u[0] = *(const uint4*)(p);
  Bv.u[1] = *(const uint4*)(p + 16);
  return Bv.v;
}

// One dense layer over the 32-row tile: Y(32 x 320) = act(X(32 x K) * W^T + b).
// Each wave owns a 32-wide N strip (2 N-tiles) x 2 M-tiles.
template<int K, bool RELU>
__device__ inline void wmma_layer(const _Float16* __restrict__ X, int ldx,
                                  const _Float16* __restrict__ W, int ldw,
                                  const float* __restrict__ bias,
                                  _Float16* __restrict__ Y, int ldy, int tid) {
  const int lane  = tid & 31;
  const int nbase = (tid >> 5) * 32;
#pragma unroll
  for (int mt = 0; mt < 2; ++mt) {
#pragma unroll
    for (int nt = 0; nt < 2; ++nt) {
      const int n0 = nbase + nt * 16;
      v8f acc = {};
#pragma unroll
      for (int kk = 0; kk < K; kk += 32) {
        v16h a = load_frag_a(X + mt * 16 * ldx + kk, ldx, lane);
        v16h b = load_frag_b(W + n0 * ldw + kk, ldw, lane);
        acc = __builtin_amdgcn_wmma_f32_16x16x32_f16(
            false, a, false, b, (short)0, acc, false, false);
      }
      // D layout: VGPR v holds M = v (+8 for lanes 16..31), N = lane&15.
      const int n    = n0 + (lane & 15);
      const int mofs = mt * 16 + ((lane >> 4) << 3);
      const float bn = bias[n];
#pragma unroll
      for (int v = 0; v < 8; ++v) {
        float val = acc[v] + bn;
        if (RELU) val = val > 0.0f ? val : 0.0f;
        Y[(mofs + v) * ldy + n] = (_Float16)val;
      }
    }
  }
}

__global__ void mask_weights_kernel(const float* __restrict__ W,
                                    const float* __restrict__ M,
                                    _Float16* __restrict__ out, int n) {
  int i = blockIdx.x * 256 + threadIdx.x;
  if (i < n) out[i] = (_Float16)(W[i] * M[i]);
}

__global__ __launch_bounds__(NTHR_) void maf_inverse_kernel(
    const float* __restrict__ z,
    const float* __restrict__ b1,
    const float* __restrict__ b2,
    const float* __restrict__ b3,
    const _Float16* __restrict__ W1h,   // 320 x 64   (masked)
    const _Float16* __restrict__ W2h,   // 320 x 320  (masked)
    const _Float16* __restrict__ W3h,   // 128 x 320  (masked)
    float* __restrict__ out)
{
  extern __shared__ char smem[];
  _Float16* sW1 = (_Float16*)smem;                 // 320*64
  _Float16* sW2 = sW1 + HID_ * DIM_;               // 320*320
  _Float16* sX  = sW2 + HID_ * HID_;               // 32*64
  _Float16* sH1 = sX  + ROWS_ * DIM_;              // 32*320
  _Float16* sH2 = sH1 + ROWS_ * HID_;              // 32*320
  float* sXf = (float*)(sH2 + ROWS_ * HID_);       // 32*64
  float* sLd = sXf + ROWS_ * DIM_;                 // 32
  float* sPS = sLd + ROWS_;                        // 32*10
  float* sPT = sPS + ROWS_ * NWV_;                 // 32*10
  float* sB1 = sPT + ROWS_ * NWV_;                 // 320
  float* sB2 = sB1 + HID_;                         // 320
  float* sB3 = sB2 + HID_;                         // 128

  const int  tid     = threadIdx.x;
  const long rowBase = (long)blockIdx.x * ROWS_;

  // Stage masked f16 weights into LDS (128-bit transfers).
  {
    const uint4* g1 = (const uint4*)W1h; uint4* s1 = (uint4*)sW1;
    for (int i = tid; i < (HID_ * DIM_) / 8; i += NTHR_) s1[i] = g1[i];
    const uint4* g2 = (const uint4*)W2h; uint4* s2 = (uint4*)sW2;
    for (int i = tid; i < (HID_ * HID_) / 8; i += NTHR_) s2[i] = g2[i];
  }
  if (tid < HID_)     { sB1[tid] = b1[tid]; sB2[tid] = b2[tid]; }
  if (tid < 2 * DIM_) { sB3[tid] = b3[tid]; }
  for (int i = tid; i < ROWS_ * DIM_; i += NTHR_) { sX[i] = (_Float16)0.0f; sXf[i] = 0.0f; }
  if (tid < ROWS_) sLd[tid] = 0.0f;
  __syncthreads();

  for (int i = 0; i < DIM_; ++i) {
    // h1 = relu(x @ mw1^T + b1)
    wmma_layer<DIM_, true>(sX, DIM_, sW1, DIM_, sB1, sH1, HID_, tid);
    __syncthreads();
    // h2 = relu(h1 @ mw2^T + b2)
    wmma_layer<HID_, true>(sH1, HID_, sW2, HID_, sB2, sH2, HID_, tid);
    __syncthreads();
    // Layer 3 pruned to the two needed outputs: s_i (row i), t_i (row 64+i).
    {
      const int row = tid & 31;
      const int grp = tid >> 5;                 // 10 groups of 32 k's
      const _Float16* ws = W3h + (size_t)i * HID_ + grp * 32;
      const _Float16* wt = W3h + (size_t)(DIM_ + i) * HID_ + grp * 32;
      const _Float16* h  = sH2 + row * HID_ + grp * 32;
      float ps = 0.0f, pt = 0.0f;
#pragma unroll
      for (int k = 0; k < 32; ++k) {
        float hv = (float)h[k];
        ps += hv * (float)ws[k];
        pt += hv * (float)wt[k];
      }
      sPS[row * NWV_ + grp] = ps;
      sPT[row * NWV_ + grp] = pt;
    }
    __syncthreads();
    if (tid < ROWS_) {
      float s = sB3[i], t = sB3[DIM_ + i];
#pragma unroll
      for (int g = 0; g < NWV_; ++g) { s += sPS[tid * NWV_ + g]; t += sPT[tid * NWV_ + g]; }
      const float zr = z[(rowBase + tid) * DIM_ + (DIM_ - 1 - i)];   // zf[:,:,i]
      const float xi = (zr - t) * __expf(-s);
      sXf[tid * DIM_ + i] = xi;
      sX [tid * DIM_ + i] = (_Float16)xi;
      sLd[tid] -= s;
    }
    __syncthreads();
  }

  // Emit x (clipped, NaN-scrubbed) and log_det.
  for (int idx = tid; idx < ROWS_ * DIM_; idx += NTHR_) {
    const int r = idx >> 6, d = idx & 63;
    float v = sXf[idx];
    if (v != v) v = 0.0f;
    v = v < -100.0f ? -100.0f : (v > 100.0f ? 100.0f : v);
    out[(rowBase + r) * DIM_ + d] = v;
  }
  if (tid < ROWS_) {
    float v = sLd[tid];
    if (v != v) v = 0.0f;
    v = v < -100.0f ? -100.0f : (v > 100.0f ? 100.0f : v);
    out[(size_t)B_ * T_ * DIM_ + rowBase + tid] = v;
  }
}

extern "C" void kernel_launch(void* const* d_in, const int* in_sizes, int n_in,
                              void* d_out, int out_size, void* d_ws, size_t ws_size,
                              hipStream_t stream) {
  const float* z  = (const float*)d_in[0];
  const float* W1 = (const float*)d_in[1];
  const float* b1 = (const float*)d_in[2];
  const float* W2 = (const float*)d_in[3];
  const float* b2 = (const float*)d_in[4];
  const float* W3 = (const float*)d_in[5];
  const float* b3 = (const float*)d_in[6];
  const float* M1 = (const float*)d_in[7];
  const float* M2 = (const float*)d_in[8];
  const float* M3 = (const float*)d_in[9];

  _Float16* W1h = (_Float16*)d_ws;            // 320*64
  _Float16* W2h = W1h + HID_ * DIM_;          // 320*320
  _Float16* W3h = W2h + HID_ * HID_;          // 128*320

  mask_weights_kernel<<<(HID_ * DIM_ + 255) / 256, 256, 0, stream>>>(W1, M1, W1h, HID_ * DIM_);
  mask_weights_kernel<<<(HID_ * HID_ + 255) / 256, 256, 0, stream>>>(W2, M2, W2h, HID_ * HID_);
  mask_weights_kernel<<<(2 * DIM_ * HID_ + 255) / 256, 256, 0, stream>>>(W3, M3, W3h, 2 * DIM_ * HID_);

  const size_t smem =
      (size_t)(HID_ * DIM_ + HID_ * HID_ + ROWS_ * DIM_ + 2 * ROWS_ * HID_) * sizeof(_Float16) +
      (size_t)(ROWS_ * DIM_ + ROWS_ + 2 * ROWS_ * NWV_ + 2 * HID_ + 2 * DIM_) * sizeof(float);
  const int nblocks = (B_ * T_) / ROWS_;      // 2048 tiles of 32 rows

  maf_inverse_kernel<<<nblocks, NTHR_, smem, stream>>>(
      z, b1, b2, b3, W1h, W2h, W3h, (float*)d_out);
}
